// ProbAttDecoder_31052613550088
// MI455X (gfx1250) — matrix operand order repaired and checked by reading, not compiled, not run on hardware
//
#include <hip/hip_runtime.h>
#include <hip/hip_bf16.h>
#include <math.h>

typedef __attribute__((ext_vector_type(16))) __bf16 v16bf;
typedef __attribute__((ext_vector_type(8)))  __bf16 v8bf;
typedef __attribute__((ext_vector_type(8)))  float  v8f;

constexpr int cB = 8, cL = 1024, cLE = 2048, cDM = 512, cNH = 8, cDH = 64;
constexpr int cDFF = 2048, cLAY = 2, cU = 35, cUP = 48;
constexpr int cBH = cB * cNH;
constexpr float cSCALE = 0.125f;   // 1/sqrt(64)

// ---------- helpers ----------
__device__ __forceinline__ __bf16 f2bf(float f) { return (__bf16)f; }

// A-fragment (16x32 MxK, 16-bit): lane holds row (lane&15); element e -> K offset
__device__ __forceinline__ int a_k(int e, int half) {
  return ((e < 8) ? e : (e + 8)) + 8 * half;
}

__device__ __forceinline__ v8f wmma_bf(v16bf a, v16bf b, v8f c) {
  return __builtin_amdgcn_wmma_f32_16x16x32_bf16(false, a, false, b, (short)0, c,
                                                 false, false);
}

__device__ __forceinline__ v16bf cat8(v8bf lo, v8bf hi) {
  return __builtin_shufflevector(lo, hi, 0, 1, 2, 3, 4, 5, 6, 7,
                                 8, 9, 10, 11, 12, 13, 14, 15);
}

// ASYNCcnt-tracked 16B global->LDS copy (per lane). GV addressing, SADDR=off.
__device__ __forceinline__ void async_b128(unsigned lds_off, const __bf16* g) {
  unsigned long long ga = (unsigned long long)(size_t)g;
  asm volatile("global_load_async_to_lds_b128 %0, %1, off"
               :: "v"(lds_off), "v"(ga) : "memory");
}
__device__ __forceinline__ void wait_async0() {
  asm volatile("s_wait_asynccnt 0x0" ::: "memory");
}
__device__ __forceinline__ unsigned lds_off_of(const void* p) {
  return (unsigned)(size_t)p;   // low 32 bits of flat shared address = LDS offset
}

// ---------- f32 -> bf16 bulk conversion (8 elems/thread) ----------
__global__ void cvt_bf16_kernel(const float* __restrict__ in, __bf16* __restrict__ out,
                                long n) {
  long i = ((long)blockIdx.x * blockDim.x + threadIdx.x) * 8;
  if (i >= n) return;
  float4 a = *(const float4*)(in + i);
  float4 b = *(const float4*)(in + i + 4);
  v8bf p = { f2bf(a.x), f2bf(a.y), f2bf(a.z), f2bf(a.w),
             f2bf(b.x), f2bf(b.y), f2bf(b.z), f2bf(b.w) };
  *(v8bf*)(out + i) = p;
}

// ---------- LDS-tiled GEMM on bf16 operands, async-copy double-buffered ----------
// C[M,N] = A[M,K] @ W[N,K]^T + bias, optional exact GELU.
// block = 256 threads (8 waves); C tile 128x64; k-step 32.
constexpr int BM = 128, BN = 64, BK = 32;

__global__ void gemm_kernel(const __bf16* __restrict__ A, int lda,
                            const __bf16* __restrict__ W, int ldw,
                            const float* __restrict__ bias,
                            float* __restrict__ C, int ldc,
                            int M, int N, int K, int act) {
  __shared__ alignas(16) __bf16 lA[2][BM * BK];
  __shared__ alignas(16) __bf16 lB[2][BN * BK];
  const int t = threadIdx.x;
  const int lane = t & 31, w = t >> 5;
  const int mh = lane & 15, half = lane >> 4;
  const int nbm = M / BM;
  const int bm = blockIdx.x % nbm, bn = blockIdx.x / nbm;
  const int wm = w & 3, wn = w >> 2;            // 4x2 wave grid -> 32x32 per wave

  // per-thread staging slices
  const __bf16* gA = A + (long)(bm * BM + (t >> 1)) * lda + (t & 1) * 16;
  const __bf16* gB = W + (long)(bn * BN + (t >> 2)) * ldw + (t & 3) * 8;
  const int sAoff = (t >> 1) * BK + (t & 1) * 16;
  const int sBoff = (t >> 2) * BK + (t & 3) * 8;

  v8f acc[2][2];
#pragma unroll
  for (int i = 0; i < 2; ++i)
#pragma unroll
    for (int j = 0; j < 2; ++j) acc[i][j] = {};

  // prologue: stage k0 = 0 into buffer 0
  async_b128(lds_off_of(&lA[0][sAoff]), gA);
  async_b128(lds_off_of(&lA[0][sAoff + 8]), gA + 8);
  async_b128(lds_off_of(&lB[0][sBoff]), gB);

  int buf = 0;
  for (int k0 = 0; k0 < K; k0 += BK, buf ^= 1) {
    wait_async0();       // this wave's copies for `buf` have landed in LDS
    __syncthreads();     // all waves' copies landed; all done reading buf^1
    if (k0 + BK < K) {   // stage next tile into the other buffer
      async_b128(lds_off_of(&lA[buf ^ 1][sAoff]), gA + k0 + BK);
      async_b128(lds_off_of(&lA[buf ^ 1][sAoff + 8]), gA + k0 + BK + 8);
      async_b128(lds_off_of(&lB[buf ^ 1][sBoff]), gB + k0 + BK);
    }
    v16bf af[2], bfr[2];
#pragma unroll
    for (int tm = 0; tm < 2; ++tm) {
      int row = wm * 32 + tm * 16 + mh;
      v8bf r1 = *(const v8bf*)&lA[buf][row * BK + 8 * half];
      v8bf r2 = *(const v8bf*)&lA[buf][row * BK + 16 + 8 * half];
      af[tm] = cat8(r1, r2);
    }
#pragma unroll
    for (int tn = 0; tn < 2; ++tn) {
      int nrow = wn * 32 + tn * 16 + mh;
      v8bf r1 = *(const v8bf*)&lB[buf][nrow * BK + 8 * half];
      v8bf r2 = *(const v8bf*)&lB[buf][nrow * BK + 16 + 8 * half];
      bfr[tn] = cat8(r1, r2);
    }
#pragma unroll
    for (int tm = 0; tm < 2; ++tm)
#pragma unroll
      for (int tn = 0; tn < 2; ++tn)
        acc[tm][tn] = wmma_bf(af[tm], bfr[tn], acc[tm][tn]);
  }

#pragma unroll
  for (int tm = 0; tm < 2; ++tm)
#pragma unroll
    for (int tn = 0; tn < 2; ++tn) {
      int col = bn * BN + wn * 32 + tn * 16 + mh;
      float bb = bias ? bias[col] : 0.f;
#pragma unroll
      for (int r = 0; r < 8; ++r) {
        int row = bm * BM + wm * 32 + tm * 16 + r + 8 * half;
        float v = acc[tm][tn][r] + bb;
        if (act) v = 0.5f * v * (1.f + erff(v * 0.70710678118f));
        C[(long)row * ldc + col] = v;
      }
    }
}

// ---------- LayerNorm(x + y) * g + b (y optional); in-place safe ----------
__global__ void ln_kernel(const float* __restrict__ x, const float* __restrict__ y,
                          const float* __restrict__ g, const float* __restrict__ b,
                          float* __restrict__ out, int D) {
  __shared__ float sh0[256], sh1[256];
  long row = blockIdx.x;
  const float* xr = x + row * D;
  const float* yr = y ? y + row * D : nullptr;
  float s = 0.f, s2 = 0.f;
  for (int i = threadIdx.x; i < D; i += blockDim.x) {
    float v = xr[i] + (yr ? yr[i] : 0.f);
    s += v; s2 += v * v;
  }
  sh0[threadIdx.x] = s; sh1[threadIdx.x] = s2;
  __syncthreads();
  for (int st = blockDim.x >> 1; st > 0; st >>= 1) {
    if ((int)threadIdx.x < st) { sh0[threadIdx.x] += sh0[threadIdx.x + st];
                                 sh1[threadIdx.x] += sh1[threadIdx.x + st]; }
    __syncthreads();
  }
  float mean = sh0[0] / D;
  float var  = sh1[0] / D - mean * mean;
  float inv  = rsqrtf(var + 1e-5f);
  for (int i = threadIdx.x; i < D; i += blockDim.x) {
    float v = xr[i] + (yr ? yr[i] : 0.f);
    out[row * D + i] = (v - mean) * inv * g[i] + b[i];
  }
}

// ---------- M[b,h,l] = max_u(q.Ksamp) - sum_u(q.Ksamp)/L : one wave per (b,h,l) ----------
__global__ void probm_kernel(const float* __restrict__ Q, const float* __restrict__ K,
                             const int* __restrict__ samp, float* __restrict__ Mout) {
  int wpb = blockDim.x >> 5;
  int gw = blockIdx.x * wpb + (threadIdx.x >> 5);
  if (gw >= cBH * cL) return;
  int lane = threadIdx.x & 31;
  int l = gw % cL, bh = gw / cL;
  int b = bh >> 3, h = bh & 7;
  const float* qr = Q + (long)(b * cL + l) * cDM + h * cDH;
  float q0 = qr[lane], q1 = qr[lane + 32];
  float mx = -1e30f, sm = 0.f;
  for (int u = 0; u < cU; ++u) {
    int j = samp[l * cU + u];
    const float* kr = K + (long)(b * cL + j) * cDM + h * cDH;
    float p = q0 * kr[lane] + q1 * kr[lane + 32];
    p += __shfl_xor(p, 16, 32); p += __shfl_xor(p, 8, 32);
    p += __shfl_xor(p, 4, 32);  p += __shfl_xor(p, 2, 32);
    p += __shfl_xor(p, 1, 32);
    mx = fmaxf(mx, p); sm += p;
  }
  if (lane == 0) Mout[(long)bh * cL + l] = mx - sm * (1.f / (float)cL);
}

// ---------- top-U selection per (b,h) ----------
__global__ void topk_kernel(const float* __restrict__ Mv, int* __restrict__ idx) {
  __shared__ float vals[cL];
  __shared__ float rv[256];
  __shared__ int   ri[256];
  int bh = blockIdx.x;
  const float* m = Mv + (long)bh * cL;
  for (int i = threadIdx.x; i < cL; i += blockDim.x) vals[i] = m[i];
  __syncthreads();
  for (int t = 0; t < cU; ++t) {
    float best = -1e30f; int bi = 0;
    for (int i = threadIdx.x; i < cL; i += blockDim.x)
      if (vals[i] > best) { best = vals[i]; bi = i; }
    rv[threadIdx.x] = best; ri[threadIdx.x] = bi;
    __syncthreads();
    for (int s = 128; s > 0; s >>= 1) {
      if ((int)threadIdx.x < s && rv[threadIdx.x + s] > rv[threadIdx.x]) {
        rv[threadIdx.x] = rv[threadIdx.x + s]; ri[threadIdx.x] = ri[threadIdx.x + s];
      }
      __syncthreads();
    }
    if (threadIdx.x == 0) { idx[bh * cU + t] = ri[0]; vals[ri[0]] = -1e30f; }
    __syncthreads();
  }
}

// ---------- scores[bh,u,l] = scale * Qr . K^T with causal mask (WMMA) ----------
__global__ void qr_scores_kernel(const float* __restrict__ Q, const float* __restrict__ K,
                                 const int* __restrict__ idx, float* __restrict__ sc) {
  int wpb = blockDim.x >> 5;
  int gw = blockIdx.x * wpb + (threadIdx.x >> 5);
  const int total = cBH * (cUP / 16) * (cL / 16);
  if (gw >= total) return;
  int lane = threadIdx.x & 31;
  int nt = gw % (cL / 16);
  int mt = (gw / (cL / 16)) % (cUP / 16);
  int bh = gw / ((cL / 16) * (cUP / 16));
  int b = bh >> 3, h = bh & 7;
  int mh = lane & 15, half = lane >> 4;
  int u = mt * 16 + mh;
  int qidx = (u < cU) ? idx[bh * cU + u] : 0;
  const float* Ar = Q + (long)(b * cL + qidx) * cDM + h * cDH;
  const float* Kr = K + (long)(b * cL + nt * 16 + mh) * cDM + h * cDH;
  v8f acc = {};
#pragma unroll
  for (int k0 = 0; k0 < cDH; k0 += 32) {
    v16bf af, bfr;
#pragma unroll
    for (int e = 0; e < 16; e += 2) {
      float2 ta = *(const float2*)(Ar + k0 + a_k(e, half));
      af[e] = f2bf(ta.x); af[e + 1] = f2bf(ta.y);
      float2 tb = *(const float2*)(Kr + k0 + a_k(e, half));
      bfr[e] = f2bf(tb.x); bfr[e + 1] = f2bf(tb.y);
    }
    acc = wmma_bf(af, bfr, acc);
  }
  int n = nt * 16 + mh;      // key position l
#pragma unroll
  for (int r = 0; r < 8; ++r) {
    int u2 = mt * 16 + r + 8 * half;
    float v = 0.f;
    if (u2 < cU) {
      int qi = idx[bh * cU + u2];
      v = (n > qi) ? -1e30f : acc[r] * cSCALE;
    }
    sc[((long)bh * cUP + u2) * cL + n] = v;
  }
}

// ---------- row softmax over L for rows (bh, u<U) ----------
__global__ void softmax_kernel(float* __restrict__ sc, int rowlen) {
  __shared__ float red[256];
  long row = blockIdx.x;
  int u = (int)(row % cU); long bh = row / cU;
  float* p = sc + (bh * cUP + u) * (long)rowlen;
  float mx = -1e30f;
  for (int i = threadIdx.x; i < rowlen; i += blockDim.x) mx = fmaxf(mx, p[i]);
  red[threadIdx.x] = mx; __syncthreads();
  for (int s = 128; s > 0; s >>= 1) {
    if ((int)threadIdx.x < s) red[threadIdx.x] = fmaxf(red[threadIdx.x], red[threadIdx.x + s]);
    __syncthreads();
  }
  mx = red[0]; __syncthreads();
  float sm = 0.f;
  for (int i = threadIdx.x; i < rowlen; i += blockDim.x) {
    float e = __expf(p[i] - mx); p[i] = e; sm += e;
  }
  red[threadIdx.x] = sm; __syncthreads();
  for (int s = 128; s > 0; s >>= 1) {
    if ((int)threadIdx.x < s) red[threadIdx.x] += red[threadIdx.x + s];
    __syncthreads();
  }
  float inv = 1.f / red[0];
  for (int i = threadIdx.x; i < rowlen; i += blockDim.x) p[i] *= inv;
}

// ---------- upd[bh,u,d] = attn @ V (WMMA, K = L) ----------
__global__ void attnv_kernel(const float* __restrict__ sc, const float* __restrict__ V,
                             float* __restrict__ upd) {
  int wpb = blockDim.x >> 5;
  int gw = blockIdx.x * wpb + (threadIdx.x >> 5);
  const int total = cBH * (cUP / 16) * (cDH / 16);
  if (gw >= total) return;
  int lane = threadIdx.x & 31;
  int nd = gw % (cDH / 16);
  int mt = (gw / (cDH / 16)) % (cUP / 16);
  int bh = gw / ((cDH / 16) * (cUP / 16));
  int b = bh >> 3, h = bh & 7;
  int mh = lane & 15, half = lane >> 4;
  const float* Ar = sc + ((long)bh * cUP + mt * 16 + mh) * cL;
  v8f acc = {};
  for (int k0 = 0; k0 < cL; k0 += 32) {
    v16bf af, bfr;
#pragma unroll
    for (int e = 0; e < 16; e += 2) {
      float2 ta = *(const float2*)(Ar + k0 + a_k(e, half));
      af[e] = f2bf(ta.x); af[e + 1] = f2bf(ta.y);
    }
#pragma unroll
    for (int e = 0; e < 16; ++e) {
      int kb = a_k(e, half);
      bfr[e] = f2bf(V[(long)(b * cL + k0 + kb) * cDM + h * cDH + nd * 16 + mh]);
    }
    acc = wmma_bf(af, bfr, acc);
  }
  int n = nd * 16 + mh;
#pragma unroll
  for (int r = 0; r < 8; ++r)
    upd[((long)bh * cUP + mt * 16 + r + 8 * half) * cDH + n] = acc[r];
}

// ---------- context = cumsum(V over l), stored (b,h,l,d) contiguous ----------
__global__ void cumsum_kernel(const float* __restrict__ V, float* __restrict__ ctx) {
  int t = blockIdx.x * blockDim.x + threadIdx.x;
  if (t >= cBH * cDH) return;
  int d = t & 63, bh = t >> 6;
  int b = bh >> 3, h = bh & 7;
  const float* vp = V + (long)b * cL * cDM + h * cDH + d;
  float* cp = ctx + (long)bh * cL * cDH + d;
  float s = 0.f;
  for (int l = 0; l < cL; ++l) { s += vp[(long)l * cDM]; cp[(long)l * cDH] = s; }
}

// ---------- context[bh, idx[u], :] = upd[bh, u, :] ----------
__global__ void scatter_kernel(const float* __restrict__ upd, const int* __restrict__ idx,
                               float* __restrict__ ctx) {
  int t = blockIdx.x * blockDim.x + threadIdx.x;
  if (t >= cBH * cU * cDH) return;
  int d = t & 63;
  int u = (t >> 6) % cU;
  int bh = t / (cU * cDH);
  int l = idx[bh * cU + u];
  ctx[((long)bh * cL + l) * cDH + d] = upd[((long)bh * cUP + u) * cDH + d];
}

// ---------- flash-style full cross attention: one wave per (b,h,16-query tile) ----------
__global__ void cross_attn_kernel(const float* __restrict__ Q, const float* __restrict__ K,
                                  const float* __restrict__ V, float* __restrict__ O) {
  __shared__ float psh[4][16 * 32];
  int wid = threadIdx.x >> 5, lane = threadIdx.x & 31;
  int gw = blockIdx.x * 4 + wid;
  if (gw >= cBH * (cL / 16)) return;
  int qt = gw & 63, bh = gw >> 6;
  int b = bh >> 3, h = bh & 7;
  int mh = lane & 15, half = lane >> 4;
  const float* Qr = Q + (long)(b * cL + qt * 16 + mh) * cDM + h * cDH;
  v16bf aq0, aq1;
#pragma unroll
  for (int e = 0; e < 16; e += 2) {
    int ka = half * 16 + e;
    float2 t0 = *(const float2*)(Qr + ka);
    aq0[e] = f2bf(t0.x * cSCALE); aq0[e + 1] = f2bf(t0.y * cSCALE);
    float2 t1 = *(const float2*)(Qr + 32 + ka);
    aq1[e] = f2bf(t1.x * cSCALE); aq1[e + 1] = f2bf(t1.y * cSCALE);
  }
  v8f oacc[4]; oacc[0] = {}; oacc[1] = {}; oacc[2] = {}; oacc[3] = {};
  float mrow[8], lrow[8];
#pragma unroll
  for (int r = 0; r < 8; ++r) { mrow[r] = -1e30f; lrow[r] = 0.f; }

  for (int s0 = 0; s0 < cLE; s0 += 32) {
    const float* K0 = K + (long)(b * cLE + s0 + mh) * cDM + h * cDH;
    const float* K1 = K0 + 16 * cDM;
    v16bf bk;
    v8f sc0 = {}, sc1 = {};
#pragma unroll
    for (int e = 0; e < 16; e += 2) {
      float2 t = *(const float2*)(K0 + half * 16 + e);
      bk[e] = f2bf(t.x); bk[e + 1] = f2bf(t.y);
    }
    sc0 = wmma_bf(aq0, bk, sc0);
#pragma unroll
    for (int e = 0; e < 16; e += 2) {
      float2 t = *(const float2*)(K0 + 32 + half * 16 + e);
      bk[e] = f2bf(t.x); bk[e + 1] = f2bf(t.y);
    }
    sc0 = wmma_bf(aq1, bk, sc0);
#pragma unroll
    for (int e = 0; e < 16; e += 2) {
      float2 t = *(const float2*)(K1 + half * 16 + e);
      bk[e] = f2bf(t.x); bk[e + 1] = f2bf(t.y);
    }
    sc1 = wmma_bf(aq0, bk, sc1);
#pragma unroll
    for (int e = 0; e < 16; e += 2) {
      float2 t = *(const float2*)(K1 + 32 + half * 16 + e);
      bk[e] = f2bf(t.x); bk[e + 1] = f2bf(t.y);
    }
    sc1 = wmma_bf(aq1, bk, sc1);

    // online softmax over this 32-key chunk (row = r + 8*half lives in 16-lane group)
#pragma unroll
    for (int r = 0; r < 8; ++r) {
      float smax = fmaxf(sc0[r], sc1[r]);
      smax = fmaxf(smax, __shfl_xor(smax, 1, 16));
      smax = fmaxf(smax, __shfl_xor(smax, 2, 16));
      smax = fmaxf(smax, __shfl_xor(smax, 4, 16));
      smax = fmaxf(smax, __shfl_xor(smax, 8, 16));
      float mnew = fmaxf(mrow[r], smax);
      float alpha = __expf(mrow[r] - mnew);
      mrow[r] = mnew;
      float p0 = __expf(sc0[r] - mnew);
      float p1 = __expf(sc1[r] - mnew);
      float ps = p0 + p1;
      ps += __shfl_xor(ps, 1, 16); ps += __shfl_xor(ps, 2, 16);
      ps += __shfl_xor(ps, 4, 16); ps += __shfl_xor(ps, 8, 16);
      lrow[r] = lrow[r] * alpha + ps;
      oacc[0][r] *= alpha; oacc[1][r] *= alpha;
      oacc[2][r] *= alpha; oacc[3][r] *= alpha;
      int rr = r + 8 * half;
      psh[wid][rr * 32 + mh]      = p0;
      psh[wid][rr * 32 + 16 + mh] = p1;
    }
    // reload P as A-fragment (per-wave LDS slab; wave-internal DS ordering)
    v16bf pa;
#pragma unroll
    for (int e = 0; e < 16; ++e) pa[e] = f2bf(psh[wid][mh * 32 + a_k(e, half)]);
#pragma unroll
    for (int j = 0; j < 4; ++j) {
      v16bf bv;
#pragma unroll
      for (int e = 0; e < 16; ++e) {
        int kb = a_k(e, half);
        bv[e] = f2bf(V[(long)(b * cLE + s0 + kb) * cDM + h * cDH + j * 16 + mh]);
      }
      oacc[j] = wmma_bf(pa, bv, oacc[j]);
    }
  }
#pragma unroll
  for (int r = 0; r < 8; ++r) {
    float inv = 1.f / lrow[r];
    int row = qt * 16 + r + 8 * half;
    float* op = O + (long)(b * cL + row) * cDM + h * cDH;
#pragma unroll
    for (int j = 0; j < 4; ++j) op[j * 16 + mh] = oacc[j][r] * inv;
  }
}

// ---------- orchestration ----------
extern "C" void kernel_launch(void* const* d_in, const int* in_sizes, int n_in,
                              void* d_out, int out_size, void* d_ws, size_t ws_size,
                              hipStream_t stream) {
  (void)in_sizes; (void)n_in; (void)out_size; (void)ws_size;
  const float* din = (const float*)d_in[0];
  const float* enc = (const float*)d_in[1];
  const float *Wq_s = (const float*)d_in[2],  *bq_s = (const float*)d_in[3];
  const float *Wk_s = (const float*)d_in[4],  *bk_s = (const float*)d_in[5];
  const float *Wv_s = (const float*)d_in[6],  *bv_s = (const float*)d_in[7];
  const float *Wo_s = (const float*)d_in[8],  *bo_s = (const float*)d_in[9];
  const float *Wq_c = (const float*)d_in[10], *bq_c = (const float*)d_in[11];
  const float *Wk_c = (const float*)d_in[12], *bk_c = (const float*)d_in[13];
  const float *Wv_c = (const float*)d_in[14], *bv_c = (const float*)d_in[15];
  const float *Wo_c = (const float*)d_in[16], *bo_c = (const float*)d_in[17];
  const float *W1 = (const float*)d_in[18], *b1 = (const float*)d_in[19];
  const float *W2 = (const float*)d_in[20], *b2 = (const float*)d_in[21];
  const float *g1 = (const float*)d_in[22], *be1 = (const float*)d_in[23];
  const float *g2 = (const float*)d_in[24], *be2 = (const float*)d_in[25];
  const float *g3 = (const float*)d_in[26], *be3 = (const float*)d_in[27];
  const float *gf = (const float*)d_in[28], *bef = (const float*)d_in[29];
  const int* samp = (const int*)d_in[30];

  const int NTOK = cB * cL, NENC = cB * cLE;
  size_t off = 0;
  auto carveB = [&](size_t bytes) -> char* {
    char* p = (char*)d_ws + off;
    off += (bytes + 255ul) / 256ul * 256ul;
    return p;
  };
  float* X   = (float*)carveB((size_t)NTOK * cDM * 4);
  float* S1  = (float*)carveB((size_t)NTOK * cDM * 4);
  float* Qb  = (float*)carveB((size_t)NTOK * cDM * 4);
  float* Kb  = (float*)carveB((size_t)NENC * cDM * 4);
  float* Vb  = (float*)carveB((size_t)NENC * cDM * 4);
  float* FF  = (float*)carveB((size_t)NTOK * cDFF * 4);
  float* SC  = (float*)carveB((size_t)cBH * cUP * cL * 4);
  float* Mb  = (float*)carveB((size_t)cBH * cL * 4);
  float* UPD = (float*)carveB((size_t)cBH * cUP * cDH * 4);
  int*   IDX = (int*)carveB((size_t)cBH * cU * 4);
  // bf16 arenas
  const size_t SA  = (size_t)cLAY * cDM * cDM;         // attn weight slab (per tensor)
  const size_t SFF = (size_t)cLAY * cDFF * cDM;        // W1/W2 slab
  __bf16* WB    = (__bf16*)carveB((8 * SA + 2 * SFF) * 2);
  __bf16* EncBf = (__bf16*)carveB((size_t)NENC * cDM * 2);
  __bf16* Abf   = (__bf16*)carveB((size_t)NTOK * cDFF * 2);

  hipMemcpyAsync(X, din, (size_t)NTOK * cDM * sizeof(float),
                 hipMemcpyDeviceToDevice, stream);

  auto cvt = [&](const float* in, __bf16* out, long n) {
    cvt_bf16_kernel<<<(unsigned)((n / 8 + 255) / 256), 256, 0, stream>>>(in, out, n);
  };
  auto gemm = [&](const __bf16* A, int lda, const __bf16* W, int ldw, const float* bias,
                  float* C, int ldc, int M, int N, int Kd, int act) {
    int blocks = (M / BM) * (N / BN);
    gemm_kernel<<<blocks, 256, 0, stream>>>(A, lda, W, ldw, bias, C, ldc, M, N, Kd, act);
  };

  // one-time conversions: all weights + encoder output
  cvt(Wq_s, WB + 0 * SA, SA); cvt(Wk_s, WB + 1 * SA, SA);
  cvt(Wv_s, WB + 2 * SA, SA); cvt(Wo_s, WB + 3 * SA, SA);
  cvt(Wq_c, WB + 4 * SA, SA); cvt(Wk_c, WB + 5 * SA, SA);
  cvt(Wv_c, WB + 6 * SA, SA); cvt(Wo_c, WB + 7 * SA, SA);
  cvt(W1, WB + 8 * SA, SFF);  cvt(W2, WB + 8 * SA + SFF, SFF);
  cvt(enc, EncBf, (long)NENC * cDM);

  for (int l = 0; l < cLAY; ++l) {
    const size_t wo = (size_t)l * cDM * cDM, bo = (size_t)l * cDM;
    const __bf16 *wqs = WB + 0 * SA + wo, *wks = WB + 1 * SA + wo;
    const __bf16 *wvs = WB + 2 * SA + wo, *wos = WB + 3 * SA + wo;
    const __bf16 *wqc = WB + 4 * SA + wo, *wkc = WB + 5 * SA + wo;
    const __bf16 *wvc = WB + 6 * SA + wo, *woc = WB + 7 * SA + wo;
    const __bf16 *w1b = WB + 8 * SA + (size_t)l * cDFF * cDM;
    const __bf16 *w2b = WB + 8 * SA + SFF + (size_t)l * cDM * cDFF;

    // ---- ProbSparse self-attention ----
    cvt(X, Abf, (long)NTOK * cDM);
    gemm(Abf, cDM, wqs, cDM, bq_s + bo, Qb, cDM, NTOK, cDM, cDM, 0);
    gemm(Abf, cDM, wks, cDM, bk_s + bo, Kb, cDM, NTOK, cDM, cDM, 0);
    gemm(Abf, cDM, wvs, cDM, bv_s + bo, Vb, cDM, NTOK, cDM, cDM, 0);
    probm_kernel<<<(cBH * cL + 7) / 8, 256, 0, stream>>>(Qb, Kb, samp + (size_t)l * cL * cU, Mb);
    topk_kernel<<<cBH, 256, 0, stream>>>(Mb, IDX);
    qr_scores_kernel<<<(cBH * 3 * (cL / 16) + 7) / 8, 256, 0, stream>>>(Qb, Kb, IDX, SC);
    softmax_kernel<<<cBH * cU, 256, 0, stream>>>(SC, cL);
    attnv_kernel<<<(cBH * 3 * 4 + 7) / 8, 256, 0, stream>>>(SC, Vb, UPD);
    cumsum_kernel<<<(cBH * cDH + 255) / 256, 256, 0, stream>>>(Vb, S1);
    scatter_kernel<<<(cBH * cU * cDH + 255) / 256, 256, 0, stream>>>(UPD, IDX, S1);
    cvt(S1, Abf, (long)NTOK * cDM);
    gemm(Abf, cDM, wos, cDM, bo_s + bo, FF, cDM, NTOK, cDM, cDM, 0);
    ln_kernel<<<NTOK, 256, 0, stream>>>(X, FF, g1 + bo, be1 + bo, X, cDM);
    // ---- full cross-attention (flash) ----
    cvt(X, Abf, (long)NTOK * cDM);
    gemm(Abf, cDM, wqc, cDM, bq_c + bo, Qb, cDM, NTOK, cDM, cDM, 0);
    gemm(EncBf, cDM, wkc, cDM, bk_c + bo, Kb, cDM, NENC, cDM, cDM, 0);
    gemm(EncBf, cDM, wvc, cDM, bv_c + bo, Vb, cDM, NENC, cDM, cDM, 0);
    cross_attn_kernel<<<(cBH * (cL / 16)) / 4, 128, 0, stream>>>(Qb, Kb, Vb, S1);
    cvt(S1, Abf, (long)NTOK * cDM);
    gemm(Abf, cDM, woc, cDM, bo_c + bo, FF, cDM, NTOK, cDM, cDM, 0);
    ln_kernel<<<NTOK, 256, 0, stream>>>(X, FF, g2 + bo, be2 + bo, X, cDM);
    // ---- FFN ----
    cvt(X, Abf, (long)NTOK * cDM);
    gemm(Abf, cDM, w1b, cDM, b1 + (size_t)l * cDFF, FF, cDFF, NTOK, cDFF, cDM, 1);
    cvt(FF, Abf, (long)NTOK * cDFF);
    gemm(Abf, cDFF, w2b, cDFF, b2 + bo, S1, cDM, NTOK, cDM, cDFF, 0);
    ln_kernel<<<NTOK, 256, 0, stream>>>(X, S1, g3 + bo, be3 + bo, X, cDM);
  }
  ln_kernel<<<NTOK, 256, 0, stream>>>(X, nullptr, gf, bef, (float*)d_out, cDM);
}
